// FusionSelfAttention_18528488915540
// MI455X (gfx1250) — matrix operand order repaired and checked by reading, not compiled, not run on hardware
//
#include <hip/hip_runtime.h>
#include <hip/hip_bf16.h>
#include <math.h>

// Problem constants (reference: B=4, N=512, D=256)
#define BB 4
#define NN 512
#define DD 256

typedef __bf16 bf16x16 __attribute__((ext_vector_type(16)));
typedef float  f32x8   __attribute__((ext_vector_type(8)));

// ---------------------------------------------------------------------------
// Fast transcendentals: lower to v_exp_f32 / v_rcp_f32 (TRANS pipe, co-executes
// with VALU FMAs on CDNA5).
// tanh(x) = 1 - 2/(e^{2x}+1);  e^{2x} = exp2(x * 2*log2(e))
// Saturation is correct: exp2(+inf)->inf -> tanh=1; exp2(-inf)->0 -> tanh=-1.
// ---------------------------------------------------------------------------
__device__ __forceinline__ float fast_tanh(float x) {
  float e = __builtin_amdgcn_exp2f(x * 2.8853900817779268f);
  return 1.0f - 2.0f * __builtin_amdgcn_rcpf(e + 1.0f);
}
__device__ __forceinline__ float fast_exp(float x) {
  return __builtin_amdgcn_exp2f(x * 1.4426950408889634f);
}

// ---------------------------------------------------------------------------
// WMMA fragment loaders (wave32, 16x16x32 bf16, contiguous K order for A/B, ISA layout for C/D).
// A (16Mx32K): lane L: m = L&15, h = L>>4; VGPR v packs K = h*16 + 2v + {0,1}
// B (32Kx16N): lane L: n = L&15, h = L>>4; VGPR v packs K = h*16 + 2v + {0,1}
// C/D (16x16 f32): lane L: n = L&15; VGPR r -> m = r + 8*(L>>4)
// fp32 sources are converted to bf16 on the fly (v_cvt).
// ---------------------------------------------------------------------------
__device__ __forceinline__ bf16x16 load_a_rowmajor(const float* __restrict__ src,
                                                   int ld, int lane) {
  int m = lane & 15, h = lane >> 4;
  bf16x16 a;
#pragma unroll
  for (int v = 0; v < 8; ++v) {
    int k = h * 16 + 2 * v;
    const float* p = src + m * ld + k;
    a[2 * v]     = (__bf16)p[0];
    a[2 * v + 1] = (__bf16)p[1];
  }
  return a;
}

// B element (k,n) = W[n*ld + k]  (i.e. B = W^T for PyTorch Linear x@W.T)
__device__ __forceinline__ bf16x16 load_b_transposed(const float* __restrict__ src,
                                                     int ld, int lane) {
  int n = lane & 15, h = lane >> 4;
  bf16x16 b;
#pragma unroll
  for (int v = 0; v < 8; ++v) {
    int k = h * 16 + 2 * v;
    const float* p = src + n * ld + k;
    b[2 * v]     = (__bf16)p[0];
    b[2 * v + 1] = (__bf16)p[1];
  }
  return b;
}

// B element (k,n) = src[k*ld + n]  (row-major K x N source; coalesced over lanes)
__device__ __forceinline__ bf16x16 load_b_rowmajor(const float* __restrict__ src,
                                                   int ld, int lane) {
  int n = lane & 15, h = lane >> 4;
  bf16x16 b;
#pragma unroll
  for (int v = 0; v < 8; ++v) {
    int k = h * 16 + 2 * v;
    b[2 * v]     = (__bf16)src[k * ld + n];
    b[2 * v + 1] = (__bf16)src[(k + 1) * ld + n];
  }
  return b;
}

// ---------------------------------------------------------------------------
// Kernel 1: q = X @ Wa^T + ba ; k = X @ Ua^T + bu
// X is [2048, 256]. One 16x16 output tile per wave, 8 WMMAs (K=256).
// 4096 tiles total (2048 per matrix), 4 waves/block -> 1024 blocks of 128.
// ---------------------------------------------------------------------------
__global__ void proj_kernel(const float* __restrict__ X,
                            const float* __restrict__ Wq, const float* __restrict__ bq,
                            const float* __restrict__ Wk, const float* __restrict__ bk,
                            float* __restrict__ Q, float* __restrict__ K) {
  int lane = threadIdx.x & 31;
  int w = blockIdx.x * (blockDim.x >> 5) + (threadIdx.x >> 5);
  int which = w >> 11;        // 0 -> q-side, 1 -> k-side (uniform per wave)
  int t = w & 2047;
  int rt = t >> 4, ct = t & 15;
  const float* Wm = which ? Wk : Wq;
  const float* bm = which ? bk : bq;
  float* Om       = which ? K  : Q;
  int row0 = rt * 16, col0 = ct * 16;

  f32x8 c = {0.f, 0.f, 0.f, 0.f, 0.f, 0.f, 0.f, 0.f};
#pragma unroll
  for (int k0 = 0; k0 < DD; k0 += 32) {
    bf16x16 a = load_a_rowmajor(X + row0 * DD + k0, DD, lane);
    bf16x16 b = load_b_transposed(Wm + col0 * DD + k0, DD, lane);
    c = __builtin_amdgcn_wmma_f32_16x16x32_bf16(false, a, false, b,
                                                (short)0, c, false, false);
  }
  int n = lane & 15, h = lane >> 4;
  float bias = bm[col0 + n];
#pragma unroll
  for (int r = 0; r < 8; ++r) {
    int m = r + 8 * h;
    Om[(row0 + m) * DD + col0 + n] = c[r] + bias;
  }
}

// ---------------------------------------------------------------------------
// Kernel 2: wei[b,i,j] = softmax_j( Va . tanh(q[b,i,:] + k[b,j,:]) + vb ), masked.
// One block per (b,i): 256 threads, 2 j-values each. q row + Va staged in LDS.
// This is the compute wall: 268M tanh total on the TRANS pipe.
// ---------------------------------------------------------------------------
__global__ void score_softmax_kernel(const float* __restrict__ Q,
                                     const float* __restrict__ K,
                                     const int* __restrict__ mask,
                                     const float* __restrict__ Va,
                                     const float* __restrict__ Vb,
                                     float* __restrict__ Wei) {
  __shared__ float qs[DD];
  __shared__ float vas[DD];
  __shared__ float red[256];

  int b = blockIdx.y, i = blockIdx.x, t = threadIdx.x;
  int row = b * NN + i;
  qs[t]  = Q[row * DD + t];
  vas[t] = Va[t];
  __syncthreads();

  float vb = Vb[0];
  float sc[2];
#pragma unroll
  for (int jj = 0; jj < 2; ++jj) {
    int j = t + jj * 256;
    const float4* kp = (const float4*)(K + (b * NN + j) * DD);
    float acc = 0.f;
#pragma unroll 4
    for (int d4 = 0; d4 < DD / 4; ++d4) {
      float4 kv = kp[d4];
      acc += vas[4 * d4 + 0] * fast_tanh(qs[4 * d4 + 0] + kv.x);
      acc += vas[4 * d4 + 1] * fast_tanh(qs[4 * d4 + 1] + kv.y);
      acc += vas[4 * d4 + 2] * fast_tanh(qs[4 * d4 + 2] + kv.z);
      acc += vas[4 * d4 + 3] * fast_tanh(qs[4 * d4 + 3] + kv.w);
    }
    float s = acc + vb;
    if (mask[(size_t)row * NN + j] == 0) s = -INFINITY;
    sc[jj] = s;
  }

  // row max
  red[t] = fmaxf(sc[0], sc[1]);
  __syncthreads();
  for (int s = 128; s > 0; s >>= 1) {
    if (t < s) red[t] = fmaxf(red[t], red[t + s]);
    __syncthreads();
  }
  float mrow = red[0];
  __syncthreads();

  float e0 = fast_exp(sc[0] - mrow);
  float e1 = fast_exp(sc[1] - mrow);
  red[t] = e0 + e1;
  __syncthreads();
  for (int s = 128; s > 0; s >>= 1) {
    if (t < s) red[t] += red[t + s];
    __syncthreads();
  }
  float inv = __builtin_amdgcn_rcpf(red[0]);

  Wei[(size_t)row * NN + t]       = e0 * inv;
  Wei[(size_t)row * NN + t + 256] = e1 * inv;
}

// ---------------------------------------------------------------------------
// Kernel 3: Y[b] = Wei[b] @ X[b]  (512x256x512 per batch, bf16 WMMA, f32 acc).
// One 16x16 tile per wave, 16 WMMAs (K=512). 2048 tiles / 4 waves -> 512 blocks.
// ---------------------------------------------------------------------------
__global__ void av_kernel(const float* __restrict__ Wei,
                          const float* __restrict__ X,
                          float* __restrict__ Y) {
  int lane = threadIdx.x & 31;
  int w = blockIdx.x * (blockDim.x >> 5) + (threadIdx.x >> 5);
  int b = w >> 9;             // 512 tiles per batch
  int r = w & 511;
  int mt = r >> 4, nt = r & 15;
  int row0 = mt * 16, col0 = nt * 16;
  const float* Wb = Wei + (size_t)b * NN * NN;
  const float* Xb = X   + (size_t)b * NN * DD;

  f32x8 c = {0.f, 0.f, 0.f, 0.f, 0.f, 0.f, 0.f, 0.f};
#pragma unroll
  for (int k0 = 0; k0 < NN; k0 += 32) {
    bf16x16 a  = load_a_rowmajor(Wb + row0 * NN + k0, NN, lane);
    bf16x16 bb = load_b_rowmajor(Xb + k0 * DD + col0, DD, lane);
    c = __builtin_amdgcn_wmma_f32_16x16x32_bf16(false, a, false, bb,
                                                (short)0, c, false, false);
  }
  int n = lane & 15, h = lane >> 4;
#pragma unroll
  for (int rr = 0; rr < 8; ++rr) {
    int m = rr + 8 * h;
    Y[(size_t)b * NN * DD + (row0 + m) * DD + col0 + n] = c[rr];
  }
}

// ---------------------------------------------------------------------------
extern "C" void kernel_launch(void* const* d_in, const int* in_sizes, int n_in,
                              void* d_out, int out_size, void* d_ws, size_t ws_size,
                              hipStream_t stream) {
  const float* X    = (const float*)d_in[0];   // out  [B,N,D]
  const int*   mask = (const int*)  d_in[1];   // mask [B,N,N]
  const float* Wa_w = (const float*)d_in[2];   // [D,D]
  const float* Wa_b = (const float*)d_in[3];   // [D]
  const float* Ua_w = (const float*)d_in[4];   // [D,D]
  const float* Ua_b = (const float*)d_in[5];   // [D]
  const float* Va_w = (const float*)d_in[6];   // [1,D]
  const float* Va_b = (const float*)d_in[7];   // [1]

  float* ws  = (float*)d_ws;
  float* Q   = ws;                               // 2048*256 f32 (2 MB)
  float* K   = Q + (size_t)BB * NN * DD;         // 2048*256 f32 (2 MB)
  float* Wei = K + (size_t)BB * NN * DD;         // 4*512*512 f32 (4 MB)
  float* Y   = (float*)d_out;

  // 1) projections: 4096 16x16 tiles, 4 waves per 128-thread block
  proj_kernel<<<1024, 128, 0, stream>>>(X, Wa_w, Wa_b, Ua_w, Ua_b, Q, K);

  // 2) scores + softmax: one block per (b,i) query row
  dim3 g2(NN, BB);
  score_softmax_kernel<<<g2, 256, 0, stream>>>(Q, K, mask, Va_w, Va_b, Wei);

  // 3) weighted sum: 2048 16x16 tiles, 4 waves per 128-thread block
  av_kernel<<<512, 128, 0, stream>>>(Wei, X, Y);
}
